// UpBlock_40802189312118
// MI455X (gfx1250) — matrix operand order repaired and checked by reading, not compiled, not run on hardware
//
#include <hip/hip_runtime.h>
#include <hip/hip_bf16.h>

// ---------------- Problem constants ----------------
#define BATCH     4
#define T_IN      2048
#define L_SEQ     4096          // 2*T
#define IN_CH     256
#define SKIP_CH   128
#define OUT_CH    128
#define KUP       5
#define D_INNER   256
#define D_STATE   16
#define D_CONV    4
#define DT_RANK   8
#define NROWS     (BATCH * L_SEQ)     // 16384
#define FUSE_K    (IN_CH + SKIP_CH)   // 384
#define XP_NPAD   64                  // 40 padded to 64
#define NCHUNK    32                  // scan chunks per sequence
#define CHUNK_LEN (L_SEQ / NCHUNK)    // 128
#define NCHAN     (BATCH * D_INNER)   // 1024 scan channels per direction

typedef __bf16 bf16_t;
typedef __attribute__((ext_vector_type(16))) __bf16 v16bf;
typedef __attribute__((ext_vector_type(8)))  float  v8f;

union FragBF { uint4 u[2]; v16bf v; };
union AccU   { v8f v; float f[8]; };

__device__ __forceinline__ unsigned short f2bf(float f) {
    unsigned int u = __float_as_uint(f);
    u += 0x7FFFu + ((u >> 16) & 1u);
    return (unsigned short)(u >> 16);
}
__device__ __forceinline__ float bf2f(unsigned short h) {
    return __uint_as_float(((unsigned int)h) << 16);
}
__device__ __forceinline__ float silu_f(float x) {
    return x / (1.0f + __expf(-x));
}

// ---------------- WMMA GEMM: C[M,N] = A[M,K](bf16) * W[N,K]^T(bf16) (+bias) ----
// A row-major (M x K), W row-major (N x K)  -> C row-major (M x N) fp32.
// One wave computes a 16(M) x 32(N) tile.  8 waves / block stacked in M.
// grid = (N/32, M/128), block = 256.
__global__ void gemm_bf16_kernel(const unsigned short* __restrict__ A,
                                 const unsigned short* __restrict__ W,
                                 float* __restrict__ C,
                                 const float* __restrict__ bias,
                                 int N, int K)
{
    const int lane  = threadIdx.x & 31;
    const int wave  = threadIdx.x >> 5;
    const int tileM = blockIdx.y * 8 + wave;
    const int tileN = blockIdx.x;
    const int hi    = lane >> 4;        // 0: lanes 0-15, 1: lanes 16-31
    const int m     = tileM * 16 + (lane & 15);
    const int n0    = tileN * 32 + (lane & 15);

    const unsigned short* arow = A + (size_t)m * K + (hi ? 16 : 0);
    const unsigned short* wr0  = W + (size_t)n0 * K + (hi ? 16 : 0);
    const unsigned short* wr1  = wr0 + (size_t)16 * K;

    v8f acc0 = {};
    v8f acc1 = {};
    for (int k0 = 0; k0 < K; k0 += 32) {
        __builtin_prefetch(arow + k0 + 64, 0, 0);
        FragBF a, b0, b1;
        a.u[0]  = *(const uint4*)(arow + k0);
        a.u[1]  = *(const uint4*)(arow + k0 + 8);
        b0.u[0] = *(const uint4*)(wr0 + k0);
        b0.u[1] = *(const uint4*)(wr0 + k0 + 8);
        b1.u[0] = *(const uint4*)(wr1 + k0);
        b1.u[1] = *(const uint4*)(wr1 + k0 + 8);
        acc0 = __builtin_amdgcn_wmma_f32_16x16x32_bf16(
                   false, a.v, false, b0.v, (short)0, acc0, false, false);
        acc1 = __builtin_amdgcn_wmma_f32_16x16x32_bf16(
                   false, a.v, false, b1.v, (short)0, acc1, false, false);
    }

    AccU u0, u1; u0.v = acc0; u1.v = acc1;
    const int nA = tileN * 32 + (lane & 15);
    const int rowBase = tileM * 16 + (hi ? 8 : 0);
    const float bv0 = bias ? bias[nA]      : 0.0f;
    const float bv1 = bias ? bias[nA + 16] : 0.0f;
#pragma unroll
    for (int r = 0; r < 8; ++r) {
        const size_t row = (size_t)(rowBase + r);
        C[row * N + nA]      = u0.f[r] + bv0;
        C[row * N + nA + 16] = u1.f[r] + bv1;
    }
}

// ---------------- Transposed conv (upsample x2, K=5) -> bf16 act rows -------
// act layout: [(b*L+t) * 384 + o]  (u part, o in 0..255)
__global__ void upsample_kernel(const float* __restrict__ x,
                                const float* __restrict__ up_w,
                                const float* __restrict__ up_b,
                                unsigned short* __restrict__ act)
{
    const int t = blockIdx.x * blockDim.x + threadIdx.x;   // 0..4095
    const int o = blockIdx.y;                              // 0..255
    const int b = blockIdx.z;
    const int s = t >> 1;
    const float* xb = x + (size_t)b * IN_CH * T_IN;
    float acc = up_b[o];
    if (t & 1) {
        // taps: up_w[i,o,3] @ x[s], up_w[i,o,1] @ x[s+1]
        const int okp = (s + 1 < T_IN);
        for (int i = 0; i < IN_CH; ++i) {
            const float* wr = up_w + ((size_t)i * IN_CH + o) * KUP;
            const float xs  = xb[(size_t)i * T_IN + s];
            const float xs1 = okp ? xb[(size_t)i * T_IN + s + 1] : 0.0f;
            acc += wr[3] * xs + wr[1] * xs1;
        }
    } else {
        // taps: up_w[i,o,4] @ x[s-1], up_w[i,o,2] @ x[s], up_w[i,o,0] @ x[s+1]
        const int okm = (s - 1 >= 0), okp = (s + 1 < T_IN);
        for (int i = 0; i < IN_CH; ++i) {
            const float* wr = up_w + ((size_t)i * IN_CH + o) * KUP;
            const float xm = okm ? xb[(size_t)i * T_IN + s - 1] : 0.0f;
            const float xs = xb[(size_t)i * T_IN + s];
            const float xp = okp ? xb[(size_t)i * T_IN + s + 1] : 0.0f;
            acc += wr[4] * xm + wr[2] * xs + wr[0] * xp;
        }
    }
    act[((size_t)b * L_SEQ + t) * FUSE_K + o] = f2bf(acc);
}

// skip (B,128,L) -> act[(b*L+t)*384 + 256 + c] bf16
__global__ void skip_copy_kernel(const float* __restrict__ skip,
                                 unsigned short* __restrict__ act)
{
    const int t = blockIdx.x * blockDim.x + threadIdx.x;
    const int c = blockIdx.y;
    const int b = blockIdx.z;
    const float v = skip[((size_t)b * SKIP_CH + c) * L_SEQ + t];
    act[((size_t)b * L_SEQ + t) * FUSE_K + IN_CH + c] = f2bf(v);
}

// ---------------- GroupNorm (1 group over C,L per batch) --------------------
__global__ void gn_reduce_kernel(const float* __restrict__ h,
                                 float* __restrict__ stats)
{
    __shared__ float ss[256], sq[256];
    const int b = blockIdx.x;
    const float* p = h + (size_t)b * L_SEQ * OUT_CH;
    float s = 0.f, q = 0.f;
    for (int i = threadIdx.x; i < L_SEQ * OUT_CH; i += 256) {
        const float v = p[i];
        s += v; q += v * v;
    }
    ss[threadIdx.x] = s; sq[threadIdx.x] = q;
    __syncthreads();
    for (int st = 128; st > 0; st >>= 1) {
        if ((int)threadIdx.x < st) {
            ss[threadIdx.x] += ss[threadIdx.x + st];
            sq[threadIdx.x] += sq[threadIdx.x + st];
        }
        __syncthreads();
    }
    if (threadIdx.x == 0) {
        const float nrm = 1.0f / (float)(L_SEQ * OUT_CH);
        const float mean = ss[0] * nrm;
        const float var  = sq[0] * nrm - mean * mean;
        stats[b * 2]     = mean;
        stats[b * 2 + 1] = rsqrtf(var + 1e-5f);
    }
}

__global__ void gn_apply_kernel(float* __restrict__ h,
                                const float* __restrict__ stats,
                                const float* __restrict__ gw,
                                const float* __restrict__ gb,
                                const float* __restrict__ pa)
{
    const size_t idx = (size_t)blockIdx.x * blockDim.x + threadIdx.x;
    const int c = (int)(idx & 127);
    const int b = (int)(idx >> 19);          // / (4096*128)
    const float mean = stats[b * 2], rstd = stats[b * 2 + 1];
    float v = (h[idx] - mean) * rstd * gw[c] + gb[c];
    h[idx] = (v >= 0.0f) ? v : pa[0] * v;
}

// ---------------- LayerNorm -> bf16 xn (fwd + time-flipped bwd) -------------
__global__ void ln_kernel(const float* __restrict__ h,
                          const float* __restrict__ lw,
                          const float* __restrict__ lb,
                          unsigned short* __restrict__ xnf,
                          unsigned short* __restrict__ xnb)
{
    __shared__ float sm[4], sv[4];
    const int row = blockIdx.x;              // b*L + l
    const int c   = threadIdx.x;             // 0..127
    const float v = h[(size_t)row * OUT_CH + c];
    float s = v;
    for (int off = 16; off; off >>= 1) s += __shfl_xor(s, off, 32);
    if ((threadIdx.x & 31) == 0) sm[threadIdx.x >> 5] = s;
    __syncthreads();
    const float mean = (sm[0] + sm[1] + sm[2] + sm[3]) * (1.0f / OUT_CH);
    const float d = v - mean;
    float q = d * d;
    for (int off = 16; off; off >>= 1) q += __shfl_xor(q, off, 32);
    if ((threadIdx.x & 31) == 0) sv[threadIdx.x >> 5] = q;
    __syncthreads();
    const float rstd = rsqrtf((sv[0] + sv[1] + sv[2] + sv[3]) * (1.0f / OUT_CH) + 1e-5f);
    const unsigned short bh = f2bf(d * rstd * lw[c] + lb[c]);
    const int b = row >> 12, l = row & (L_SEQ - 1);
    xnf[(size_t)row * OUT_CH + c] = bh;
    xnb[(((size_t)b << 12) + (L_SEQ - 1 - l)) * OUT_CH + c] = bh;
}

// ---------------- depthwise causal conv + SiLU, and SiLU gate ---------------
// xz rows of 512: [:256]=conv input, [256:]=gate z.
__global__ void conv_silu_kernel(const float* __restrict__ xz,
                                 const float* __restrict__ cw,
                                 const float* __restrict__ cb,
                                 float* __restrict__ xif,
                                 unsigned short* __restrict__ xibf,
                                 unsigned short* __restrict__ szbf)
{
    const size_t idx = (size_t)blockIdx.x * blockDim.x + threadIdx.x;
    const int e = (int)(idx & 511);
    const size_t row = idx >> 9;             // b*L + l
    if (e < D_INNER) {
        const int d = e;
        const int b = (int)(row >> 12);
        const int l = (int)(row & (L_SEQ - 1));
        float acc = cb[d];
#pragma unroll
        for (int j = 0; j < D_CONV; ++j) {
            const int ll = l - (D_CONV - 1) + j;
            if (ll >= 0)
                acc += cw[d * D_CONV + j] *
                       xz[((((size_t)b << 12) + ll) << 9) + d];
        }
        const float sv = silu_f(acc);
        xif[row * D_INNER + d]  = sv;
        xibf[row * D_INNER + d] = f2bf(sv);
    } else {
        const int d = e - D_INNER;
        szbf[row * D_INNER + d] = f2bf(silu_f(xz[(row << 9) + e]));
    }
}

// ---------------- dt = softplus(dbl[:,:8] @ dw^T + db) ----------------------
__global__ void dt_kernel(const float* __restrict__ dbl,
                          const float* __restrict__ dw,
                          const float* __restrict__ db,
                          float* __restrict__ dt)
{
    const size_t idx = (size_t)blockIdx.x * blockDim.x + threadIdx.x;
    const int d = (int)(idx & 255);
    const size_t row = idx >> 8;
    const float* r = dbl + (row << 6);       // stride 64
    float a = db[d];
#pragma unroll
    for (int j = 0; j < DT_RANK; ++j) a += r[j] * dw[d * DT_RANK + j];
    dt[idx] = (a > 20.0f) ? a : log1pf(__expf(a));
}

// ---------------- chunked selective scan ------------------------------------
// Recurrence per (b,d,n): h_l = a_l*h_{l-1} + u_l,  a_l = exp(dt*A), u_l = dt*B*xi.
// Split L into NCHUNK chunks; 16 lanes = 16 states, 2 (channel,chunk) units/wave.
// Pass 1: per-chunk (P = prod a, S = state from 0).
__global__ void scan_part1_kernel(const float* __restrict__ dt,
                                  const float* __restrict__ dbl,
                                  const float* __restrict__ xif,
                                  const float* __restrict__ Aneg,
                                  float* __restrict__ Pb,
                                  float* __restrict__ Sb)
{
    const int lane = threadIdx.x & 31;
    const int wv   = (int)((blockIdx.x * blockDim.x + threadIdx.x) >> 5);
    const int unit  = wv * 2 + (lane >> 4);    // 0 .. NCHAN*NCHUNK-1
    const int chunk = unit & (NCHUNK - 1);
    const int ch    = unit >> 5;               // 0..1023  (b*256 + d)
    const int n     = lane & 15;
    const int b     = ch >> 8;
    const int d     = ch & 255;
    const float an  = Aneg[d * D_STATE + n];
    const size_t rbase = ((size_t)b << 12) + (size_t)chunk * CHUNK_LEN;
    float P = 1.0f, S = 0.0f;
    for (int l = 0; l < CHUNK_LEN; ++l) {
        const size_t row = rbase + l;
        const float dtv = dt[(row << 8) + d];
        const float Bv  = dbl[(row << 6) + DT_RANK + n];
        const float xiv = xif[(row << 8) + d];
        const float a   = __expf(dtv * an);
        S = a * S + dtv * Bv * xiv;
        P *= a;
    }
    const size_t o = ((size_t)(ch * D_STATE + n)) * NCHUNK + chunk;
    Pb[o] = P;
    Sb[o] = S;
}

// Pass 2 (fix-up): sequential over the NCHUNK chunk summaries per (ch,n).
__global__ void scan_fix_kernel(const float* __restrict__ Pb,
                                const float* __restrict__ Sb,
                                float* __restrict__ Init)
{
    const int t = blockIdx.x * blockDim.x + threadIdx.x;   // 0..NCHAN*16-1
    const size_t base = (size_t)t * NCHUNK;
    float carry = 0.0f;
#pragma unroll
    for (int c = 0; c < NCHUNK; ++c) {
        Init[base + c] = carry;
        carry = Pb[base + c] * carry + Sb[base + c];
    }
}

// Pass 3: re-run chunks from corrected initial state, emit gated output G.
__global__ void scan_part2_kernel(const float* __restrict__ dt,
                                  const float* __restrict__ dbl,
                                  const float* __restrict__ xif,
                                  const unsigned short* __restrict__ szbf,
                                  const float* __restrict__ Aneg,
                                  const float* __restrict__ Dp,
                                  const float* __restrict__ Init,
                                  unsigned short* __restrict__ G)
{
    const int lane = threadIdx.x & 31;
    const int wv   = (int)((blockIdx.x * blockDim.x + threadIdx.x) >> 5);
    const int unit  = wv * 2 + (lane >> 4);
    const int chunk = unit & (NCHUNK - 1);
    const int ch    = unit >> 5;
    const int n     = lane & 15;
    const int b     = ch >> 8;
    const int d     = ch & 255;
    const float an  = Aneg[d * D_STATE + n];
    const float dpd = Dp[d];
    const size_t rbase = ((size_t)b << 12) + (size_t)chunk * CHUNK_LEN;
    float hst = Init[((size_t)(ch * D_STATE + n)) * NCHUNK + chunk];
    for (int l = 0; l < CHUNK_LEN; ++l) {
        const size_t row = rbase + l;
        const float dtv = dt[(row << 8) + d];
        const float Bv  = dbl[(row << 6) + DT_RANK + n];
        const float Cv  = dbl[(row << 6) + DT_RANK + D_STATE + n];
        const float xiv = xif[(row << 8) + d];
        const float a   = __expf(dtv * an);
        hst = a * hst + dtv * Bv * xiv;
        float y = hst * Cv;
        y += __shfl_xor(y, 1, 16);
        y += __shfl_xor(y, 2, 16);
        y += __shfl_xor(y, 4, 16);
        y += __shfl_xor(y, 8, 16);
        if (n == 0) {
            const float g = (y + dpd * xiv) * bf2f(szbf[(row << 8) + d]);
            G[(row << 8) + d] = f2bf(g);
        }
    }
}

// ---------------- residual: h += outF + flip(outB) --------------------------
__global__ void residual_add_kernel(float* __restrict__ h,
                                    const float* __restrict__ outF,
                                    const float* __restrict__ outB)
{
    const size_t idx = (size_t)blockIdx.x * blockDim.x + threadIdx.x;
    const int c = (int)(idx & 127);
    const size_t row = idx >> 7;
    const int b = (int)(row >> 12);
    const int l = (int)(row & (L_SEQ - 1));
    const size_t frow = ((size_t)b << 12) + (L_SEQ - 1 - l);
    h[idx] += outF[idx] + outB[(frow << 7) + c];
}

// ---------------- final (B,L,C) -> (B,C,L) ----------------------------------
__global__ void out_transpose_kernel(const float* __restrict__ h,
                                     float* __restrict__ out)
{
    const size_t idx = (size_t)blockIdx.x * blockDim.x + threadIdx.x; // (b,c,l)
    const int l = (int)(idx & (L_SEQ - 1));
    const size_t t = idx >> 12;
    const int c = (int)(t & 127);
    const int b = (int)(t >> 7);
    out[idx] = h[((((size_t)b << 12) + l) << 7) + c];
}

// ---------------- weight prep -----------------------------------------------
__global__ void f2bf_kernel(const float* __restrict__ src,
                            unsigned short* __restrict__ dst, int n)
{
    const int i = blockIdx.x * blockDim.x + threadIdx.x;
    if (i < n) dst[i] = f2bf(src[i]);
}

__global__ void xproj_pad_kernel(const float* __restrict__ src,  // (4,40,256)
                                 unsigned short* __restrict__ dst) // (4,64,256)
{
    const int i = blockIdx.x * blockDim.x + threadIdx.x;   // over 4*64*256
    const int k = i & 255;
    const int r = (i >> 8) & 63;
    const int g = i >> 14;
    const float v = (r < 40) ? src[((size_t)g * 40 + r) * 256 + k] : 0.0f;
    dst[i] = f2bf(v);
}

__global__ void aprep_kernel(const float* __restrict__ alog,
                             float* __restrict__ an, int n)
{
    const int i = blockIdx.x * blockDim.x + threadIdx.x;
    if (i < n) an[i] = -__expf(alog[i]);
}

// ============================================================================
extern "C" void kernel_launch(void* const* d_in, const int* in_sizes, int n_in,
                              void* d_out, int out_size, void* d_ws, size_t ws_size,
                              hipStream_t stream)
{
    (void)in_sizes; (void)n_in; (void)out_size; (void)ws_size;
    const float* x        = (const float*)d_in[0];
    const float* skip     = (const float*)d_in[1];
    const float* up_w     = (const float*)d_in[2];
    const float* up_b     = (const float*)d_in[3];
    const float* fus_w    = (const float*)d_in[4];
    const float* fus_b    = (const float*)d_in[5];
    const float* gn_w     = (const float*)d_in[6];
    const float* gn_b     = (const float*)d_in[7];
    const float* prelu_a  = (const float*)d_in[8];
    const float* ln_w     = (const float*)d_in[9];
    const float* ln_b     = (const float*)d_in[10];
    const float* in_w     = (const float*)d_in[11];
    const float* conv_w   = (const float*)d_in[12];
    const float* conv_b   = (const float*)d_in[13];
    const float* xproj_w  = (const float*)d_in[14];
    const float* dtproj_w = (const float*)d_in[15];
    const float* dtproj_b = (const float*)d_in[16];
    const float* A_log    = (const float*)d_in[17];
    const float* Dp       = (const float*)d_in[18];
    const float* out_w    = (const float*)d_in[19];
    float* out = (float*)d_out;

    // ---- workspace arena (256B aligned slices) ----
    char* base = (char*)d_ws;
    size_t off = 0;
    auto alloc = [&](size_t bytes) -> char* {
        char* p = base + off;
        off += (bytes + 255) & ~(size_t)255;
        return p;
    };
    unsigned short* WFUS = (unsigned short*)alloc((size_t)OUT_CH * FUSE_K * 2);      // 128x384
    unsigned short* WIN  = (unsigned short*)alloc((size_t)4 * 512 * 128 * 2);
    unsigned short* WXP  = (unsigned short*)alloc((size_t)4 * XP_NPAD * 256 * 2);
    unsigned short* WOUT = (unsigned short*)alloc((size_t)4 * 128 * 256 * 2);
    float* ANEG  = (float*)alloc((size_t)4 * 256 * 16 * 4);
    float* STATS = (float*)alloc(256);
    float* H     = (float*)alloc((size_t)NROWS * OUT_CH * 4);                        // 8 MB
    unsigned short* XNF = (unsigned short*)alloc((size_t)NROWS * OUT_CH * 2);
    unsigned short* XNB = (unsigned short*)alloc((size_t)NROWS * OUT_CH * 2);
    char* R1 = alloc((size_t)NROWS * 512 * 4);                                       // 32 MB
    unsigned short* ACT = (unsigned short*)R1;           // (NROWS x 384) bf16
    float* XZ  = (float*)R1;                             // (NROWS x 512) f32
    float* DBL = (float*)R1;                             // (NROWS x 64)  f32
    float* DTB = (float*)(R1 + (size_t)NROWS * 64 * 4);  // (NROWS x 256) f32
    unsigned short* G = (unsigned short*)(R1 + (size_t)NROWS * 64 * 4
                                             + (size_t)NROWS * 256 * 4);
    float* XIF  = (float*)alloc((size_t)NROWS * D_INNER * 4);                        // 16 MB
    unsigned short* XIBF = (unsigned short*)alloc((size_t)NROWS * D_INNER * 2);
    unsigned short* SZBF = (unsigned short*)alloc((size_t)NROWS * D_INNER * 2);
    float* OUTF = (float*)alloc((size_t)NROWS * OUT_CH * 4);
    float* OUTB = (float*)alloc((size_t)NROWS * OUT_CH * 4);
    const size_t scanElems = (size_t)NCHAN * D_STATE * NCHUNK;   // 524288
    float* SCAN_P    = (float*)alloc(scanElems * 4);             // 2 MB
    float* SCAN_S    = (float*)alloc(scanElems * 4);             // 2 MB
    float* SCAN_INIT = (float*)alloc(scanElems * 4);             // 2 MB

    auto gemm = [&](const unsigned short* A, const unsigned short* W, float* C,
                    const float* bias, int N, int K) {
        dim3 grid(N / 32, NROWS / 128);
        gemm_bf16_kernel<<<grid, dim3(256), 0, stream>>>(A, W, C, bias, N, K);
    };

    // ---- weight prep ----
    f2bf_kernel<<<(OUT_CH * FUSE_K + 255) / 256, 256, 0, stream>>>(fus_w, WFUS, OUT_CH * FUSE_K);
    f2bf_kernel<<<(4 * 512 * 128 + 255) / 256, 256, 0, stream>>>(in_w, WIN, 4 * 512 * 128);
    f2bf_kernel<<<(4 * 128 * 256 + 255) / 256, 256, 0, stream>>>(out_w, WOUT, 4 * 128 * 256);
    xproj_pad_kernel<<<(4 * XP_NPAD * 256) / 256, 256, 0, stream>>>(xproj_w, WXP);
    aprep_kernel<<<(4 * 256 * 16 + 255) / 256, 256, 0, stream>>>(A_log, ANEG, 4 * 256 * 16);

    // ---- upsample + skip -> fused activation rows (bf16) ----
    upsample_kernel<<<dim3(L_SEQ / 256, IN_CH, BATCH), dim3(256), 0, stream>>>(x, up_w, up_b, ACT);
    skip_copy_kernel<<<dim3(L_SEQ / 256, SKIP_CH, BATCH), dim3(256), 0, stream>>>(skip, ACT);

    // ---- fusion GEMM: (16384 x 384) * (128 x 384)^T + bias -> H ----
    gemm(ACT, WFUS, H, fus_b, OUT_CH, FUSE_K);

    // ---- GroupNorm + PReLU ----
    gn_reduce_kernel<<<BATCH, 256, 0, stream>>>(H, STATS);
    gn_apply_kernel<<<(NROWS * OUT_CH) / 256, 256, 0, stream>>>(H, STATS, gn_w, gn_b, prelu_a);

    // ---- Mamba layers ----
    const int scanUnits  = NCHAN * NCHUNK;          // 32768 (chan,chunk) units
    const int scanBlocks = scanUnits / 2 / 8;       // 2 units/wave, 8 waves/block
    for (int i = 0; i < 2; ++i) {
        ln_kernel<<<NROWS, OUT_CH, 0, stream>>>(H, ln_w + i * OUT_CH, ln_b + i * OUT_CH, XNF, XNB);
        for (int dir = 0; dir < 2; ++dir) {
            const int g = i * 2 + dir;
            const unsigned short* xn = dir ? XNB : XNF;
            float* outbuf = dir ? OUTB : OUTF;
            // in-projection: (16384 x 128) * (512 x 128)^T -> XZ
            gemm(xn, WIN + (size_t)g * 512 * 128, XZ, nullptr, 512, 128);
            // depthwise causal conv + SiLU, gate SiLU
            conv_silu_kernel<<<(NROWS * 512) / 256, 256, 0, stream>>>(
                XZ, conv_w + (size_t)g * D_INNER * D_CONV,
                conv_b + (size_t)g * D_INNER, XIF, XIBF, SZBF);
            // x-projection: (16384 x 256) * (64 x 256)^T -> DBL
            gemm(XIBF, WXP + (size_t)g * XP_NPAD * 256, DBL, nullptr, XP_NPAD, 256);
            // dt
            dt_kernel<<<(NROWS * D_INNER) / 256, 256, 0, stream>>>(
                DBL, dtproj_w + (size_t)g * D_INNER * DT_RANK,
                dtproj_b + (size_t)g * D_INNER, DTB);
            // chunked selective scan -> gated activation G (bf16)
            const float* an = ANEG + (size_t)g * D_INNER * D_STATE;
            scan_part1_kernel<<<scanBlocks, 256, 0, stream>>>(
                DTB, DBL, XIF, an, SCAN_P, SCAN_S);
            scan_fix_kernel<<<(NCHAN * D_STATE) / 256, 256, 0, stream>>>(
                SCAN_P, SCAN_S, SCAN_INIT);
            scan_part2_kernel<<<scanBlocks, 256, 0, stream>>>(
                DTB, DBL, XIF, SZBF, an, Dp + (size_t)g * D_INNER, SCAN_INIT, G);
            // out-projection: (16384 x 256) * (128 x 256)^T -> outF/outB
            gemm(G, WOUT + (size_t)g * 128 * 256, outbuf, nullptr, OUT_CH, 256);
        }
        residual_add_kernel<<<(NROWS * OUT_CH) / 256, 256, 0, stream>>>(H, OUTF, OUTB);
    }

    // ---- (B,L,C) -> (B,C,L) ----
    out_transpose_kernel<<<(NROWS * OUT_CH) / 256, 256, 0, stream>>>(H, out);
}
